// ShadowFreeTernaryLinear_14259291422935
// MI455X (gfx1250) — matrix operand order repaired and checked by reading, not compiled, not run on hardware
//
#include <hip/hip_runtime.h>
#include <cmath>

typedef __attribute__((ext_vector_type(16))) _Float16 v16h;
typedef __attribute__((ext_vector_type(8)))  float    v8f;
typedef __attribute__((ext_vector_type(4)))  float    v4f;
typedef __attribute__((ext_vector_type(4)))  int      v4i;

// Problem constants (reference: B=32, IN=8192, OUT=8192)
constexpr int BATCH = 32;
constexpr int IN_F  = 8192;
constexpr int OUT_F = 8192;

constexpr int NBLK   = 32;              // output channels per block (2 WMMA N-tiles)
constexpr int WAVES  = 16;              // waves per block (K-split)
constexpr int THREADS = WAVES * 32;     // 512
constexpr int KSEG   = IN_F / WAVES;    // 512 K per wave
constexpr int KSTEP  = 32;              // K per WMMA

static __device__ __forceinline__ v8f wmma_f16(v16h a, v16h b, v8f c) {
    return __builtin_amdgcn_wmma_f32_16x16x32_f16(
        /*neg_a=*/false, a, /*neg_b=*/false, b,
        /*c_mod=*/(short)0, c, /*reuse_a=*/false, /*reuse_b=*/false);
}

__global__ __launch_bounds__(THREADS)
void ternary_linear_wmma(const float* __restrict__ x,
                         const float* __restrict__ log_scale,
                         const float* __restrict__ bias,
                         const int*   __restrict__ wst,
                         float*       __restrict__ out)
{
    __shared__ float red[WAVES * BATCH * NBLK];   // 16 * 1024 f32 = 64 KB

    const int tid   = threadIdx.x;
    const int wave  = tid >> 5;
    const int lane  = tid & 31;
    const int sel   = lane >> 4;     // which half-wave (K-chunk select)
    const int laneM = lane & 15;     // row within 16
    const int nbase = blockIdx.x * NBLK;

    // A rows (activations, fp32 in memory; L2-resident across blocks)
    const float* rowA0 = x + (size_t)laneM * IN_F;          // M = 0..15
    const float* rowA1 = x + (size_t)(laneM + 16) * IN_F;   // M = 16..31
    // B rows (ternary weights, row-major [OUT][IN]); lane%16 = output channel
    const int* rowB0 = wst + (size_t)(nbase + laneM) * IN_F;        // N = 0..15
    const int* rowB1 = wst + (size_t)(nbase + 16 + laneM) * IN_F;   // N = 16..31

    v8f acc00 = {}, acc01 = {}, acc10 = {}, acc11 = {};

    const int kstart = wave * KSEG;

#pragma unroll 2
    for (int kk = 0; kk < KSEG; kk += KSTEP) {
        const int k = kstart + kk;

        // ---- A tiles: fp32 -> f16, ISA A-layout (16x32, 16b elements) ----
        // lanes 0-15: K chunks [0..7] & [16..23]; lanes 16-31: [8..15] & [24..31]
        const int kA = k + 8 * sel;
        v4f a00 = *(const v4f*)(rowA0 + kA);
        v4f a01 = *(const v4f*)(rowA0 + kA + 4);
        v4f a02 = *(const v4f*)(rowA0 + kA + 16);
        v4f a03 = *(const v4f*)(rowA0 + kA + 20);
        v4f a10 = *(const v4f*)(rowA1 + kA);
        v4f a11 = *(const v4f*)(rowA1 + kA + 4);
        v4f a12 = *(const v4f*)(rowA1 + kA + 16);
        v4f a13 = *(const v4f*)(rowA1 + kA + 20);
        v16h A0, A1;
#pragma unroll
        for (int i = 0; i < 4; ++i) {
            A0[i]      = (_Float16)a00[i];
            A0[4 + i]  = (_Float16)a01[i];
            A0[8 + i]  = (_Float16)a02[i];
            A0[12 + i] = (_Float16)a03[i];
            A1[i]      = (_Float16)a10[i];
            A1[4 + i]  = (_Float16)a11[i];
            A1[8 + i]  = (_Float16)a12[i];
            A1[12 + i] = (_Float16)a13[i];
        }

        // ---- B tiles: ternary int32 -> f16 (exact), nontemporal stream ----
        // lanes 0-15: K chunks [0..7] & [16..23]; lanes 16-31: [8..15] & [24..31]
        const int kB = k + 8 * sel;
        v4i b00 = __builtin_nontemporal_load((const v4i*)(rowB0 + kB));
        v4i b01 = __builtin_nontemporal_load((const v4i*)(rowB0 + kB + 4));
        v4i b02 = __builtin_nontemporal_load((const v4i*)(rowB0 + kB + 16));
        v4i b03 = __builtin_nontemporal_load((const v4i*)(rowB0 + kB + 20));
        v4i b10 = __builtin_nontemporal_load((const v4i*)(rowB1 + kB));
        v4i b11 = __builtin_nontemporal_load((const v4i*)(rowB1 + kB + 4));
        v4i b12 = __builtin_nontemporal_load((const v4i*)(rowB1 + kB + 16));
        v4i b13 = __builtin_nontemporal_load((const v4i*)(rowB1 + kB + 20));
        v16h Bt0, Bt1;
#pragma unroll
        for (int i = 0; i < 4; ++i) {
            Bt0[i]      = (_Float16)b00[i];
            Bt0[4 + i]  = (_Float16)b01[i];
            Bt0[8 + i]  = (_Float16)b02[i];
            Bt0[12 + i] = (_Float16)b03[i];
            Bt1[i]      = (_Float16)b10[i];
            Bt1[4 + i]  = (_Float16)b11[i];
            Bt1[8 + i]  = (_Float16)b12[i];
            Bt1[12 + i] = (_Float16)b13[i];
        }

        acc00 = wmma_f16(A0, Bt0, acc00);
        acc01 = wmma_f16(A0, Bt1, acc01);
        acc10 = wmma_f16(A1, Bt0, acc10);
        acc11 = wmma_f16(A1, Bt1, acc11);
    }

    // ---- spill per-wave partials to LDS (C layout: vgpr r -> M=r+8*sel) ----
    float* myred = red + wave * (BATCH * NBLK);
#pragma unroll
    for (int r = 0; r < 8; ++r) {
        const int m0 = r + 8 * sel;
        myred[m0 * NBLK + laneM]             = acc00[r];
        myred[m0 * NBLK + 16 + laneM]        = acc01[r];
        myred[(m0 + 16) * NBLK + laneM]      = acc10[r];
        myred[(m0 + 16) * NBLK + 16 + laneM] = acc11[r];
    }
    __syncthreads();

    // ---- fixed-order K reduction (bitwise deterministic) + epilogue ----
    const int nl = tid & (NBLK - 1);     // constant per thread across j
    const int o  = nbase + nl;
    const float ls = log_scale[o];
    float sp = (ls > 20.f) ? ls : log1pf(__expf(ls));   // softplus
    sp = fmaxf(sp, 1e-4f);
    const float bo = bias[o];

#pragma unroll
    for (int j = 0; j < (BATCH * NBLK) / THREADS; ++j) {   // 2 iters
        const int idx = tid + j * THREADS;
        float s = red[idx];
#pragma unroll
        for (int w2 = 1; w2 < WAVES; ++w2)
            s += red[w2 * (BATCH * NBLK) + idx];
        const int m = idx / NBLK;
        out[(size_t)m * OUT_F + o] = fmaf(s, sp, bo);
    }
}

extern "C" void kernel_launch(void* const* d_in, const int* in_sizes, int n_in,
                              void* d_out, int out_size, void* d_ws, size_t ws_size,
                              hipStream_t stream)
{
    // setup_inputs() order: inputs, log_scale, bias, weight_state
    const float* x         = (const float*)d_in[0];
    const float* log_scale = (const float*)d_in[1];
    const float* bias      = (const float*)d_in[2];
    const int*   wst       = (const int*)d_in[3];
    float*       out       = (float*)d_out;

    dim3 grid(OUT_F / NBLK);   // 256 blocks
    dim3 block(THREADS);       // 512 threads = 16 waves (K-split)
    ternary_linear_wmma<<<grid, block, 0, stream>>>(x, log_scale, bias, wst, out);
}
